// GuardianExpert_50053548867793
// MI455X (gfx1250) — matrix operand order repaired and checked by reading, not compiled, not run on hardware
//
#include <hip/hip_runtime.h>
#include <hip/hip_bf16.h>

typedef __attribute__((ext_vector_type(16))) __bf16 v16bf;
typedef __attribute__((ext_vector_type(8)))  float  v8f;

struct EncPtrs {
    const float *w1, *b1, *g1, *be1, *m1, *v1;
    const float *w2, *b2, *g2, *be2, *m2, *v2;
};

union BU { v16bf bv; unsigned short us[16]; };

__device__ __forceinline__ unsigned short f2bf(float f) {
    unsigned int u = __builtin_bit_cast(unsigned int, f);
    u += 0x7FFFu + ((u >> 16) & 1u);           // round-to-nearest-even
    return (unsigned short)(u >> 16);
}

// K' permutation: K' = (ky*3+kx)*16 + ic  (tap-major, channel-minor).
// rowoff(q) = pixel offset of tap q = ky*PS + kx.
__host__ __device__ constexpr int rowoff(int q, int PS) {
    return (q / 3) * PS + (q % 3);
}

// ---------------------------------------------------------------------------
// Kernel 0: fold BN into weights/biases, pack conv2 weights into WMMA
// A-fragment layout (bf16) using the permuted K' index, zero feature acc.
// fws layout (floats): [0..511] featAcc (2 enc * 8 batch * 32 ch)
//   per enc e at 512 + e*512:  w1f[144] @ +0, b1f[16] @ +160, b2f[32] @ +192
// afrag (ushort) at byte 8192: per enc e*5120: [mtile(2)][kt(5)][lane(32)][16]
// ---------------------------------------------------------------------------
__global__ void guardian_prep(EncPtrs e0, EncPtrs e1, float* fws,
                              unsigned short* afrag) {
    const int t = threadIdx.x;
    for (int i = t; i < 512; i += 256) fws[i] = 0.0f;

    EncPtrs encs[2] = { e0, e1 };
    for (int e = 0; e < 2; ++e) {
        const EncPtrs P = encs[e];
        float* w1f = fws + 512 + e * 512;
        float* b1f = w1f + 160;
        float* b2f = w1f + 192;
        if (t < 16) {
            float s1 = P.g1[t] * rsqrtf(P.v1[t] + 1e-5f);
            b1f[t] = P.b1[t] * s1 + P.be1[t] - P.m1[t] * s1;
        }
        if (t < 32) {
            float s2 = P.g2[t] * rsqrtf(P.v2[t] + 1e-5f);
            b2f[t] = P.b2[t] * s2 + P.be2[t] - P.m2[t] * s2;
        }
        for (int i = t; i < 144; i += 256) {
            int oc = i / 9;
            float s1 = P.g1[oc] * rsqrtf(P.v1[oc] + 1e-5f);
            w1f[i] = P.w1[i] * s1;
        }
        unsigned short* af = afrag + e * 5120;
        for (int i = t; i < 5120; i += 256) {
            int ii    = i & 15;
            int lane  = (i >> 4) & 31;
            int kt    = (i >> 9) % 5;
            int mtile = (i >> 9) / 5;
            int m = mtile * 16 + (lane & 15);
            // contiguous K order used here (B matches): lanes 0-15 K={0..15}, lanes 16-31 K={16..31}
            int klocal = ((lane >> 4) ? 16 : 0) + ii;
            int k = kt * 32 + klocal;               // permuted K'
            unsigned short hv = 0;
            if (k < 144) {
                int q  = k >> 4;                    // tap: ky*3+kx
                int ic = k & 15;                    // input channel
                float s2 = P.g2[m] * rsqrtf(P.v2[m] + 1e-5f);
                hv = f2bf(P.w2[(m * 16 + ic) * 9 + q] * s2);
            }
            af[i] = hv;
        }
    }
}

// ---------------------------------------------------------------------------
// Kernel 1/2: per s*s block — gather, bitonic sort, conv1(BN,ReLU) in VALU,
// conv2 as im2col GEMM via v_wmma_f32_16x16x32_bf16. Channel-minor H1c
// (32B/pixel) + permuted K': a B fragment is one contiguous 32B LDS read.
// ---------------------------------------------------------------------------
template <int S>
__global__ __launch_bounds__((S == 8) ? 128 : 256)
void guardian_block_enc(const float* __restrict__ c,
                        const float* __restrict__ w1f,
                        const float* __restrict__ b1f,
                        const float* __restrict__ b2f,
                        const unsigned short* __restrict__ afrag,
                        float* __restrict__ featAcc,
                        int sidx) {
    constexpr int NTH   = (S == 8) ? 128 : 256;
    constexpr int NPIX  = S * S;
    constexpr int PS    = S + 2;
    constexpr int NTILE = NPIX / 16;       // tasks: one per N-tile of 16 pixels
    constexpr int NB    = 768 / S;

    __shared__ float                     sbuf[NPIX];
    __shared__ float                     Ppad[PS * PS];
    __shared__ alignas(32) unsigned short H1c[PS * PS * 16]; // [pixel][ic]
    __shared__ float                     blockAcc[32];
    __shared__ float                     b2s[32];
    __shared__ float                     w1s[144];
    __shared__ float                     b1s[16];

    const int t   = threadIdx.x;
    const int bid = blockIdx.x;
    const int b   = bid / (NB * NB);
    const int rem = bid - b * (NB * NB);
    const int by  = rem / NB;
    const int bx  = rem - by * NB;

    for (int i = t; i < PS * PS; i += NTH) Ppad[i] = 0.0f;
    {
        int4* hz = (int4*)H1c;
        int4  z  = {0, 0, 0, 0};
        for (int i = t; i < PS * PS * 2; i += NTH) hz[i] = z;
    }
    if (t < 32) { blockAcc[t] = 0.0f; b2s[t] = b2f[t]; }
    if (t < 16) b1s[t] = b1f[t];
    for (int i = t; i < 144; i += NTH) w1s[i] = w1f[i];

    if (t < NPIX) {
        int y = t / S, x = t - (t / S) * S;
        sbuf[t] = c[((long)b * 768 + by * S + y) * 768 + bx * S + x];
    }
    __syncthreads();

    // ascending bitonic sort of the block's pixels
    for (int k = 2; k <= NPIX; k <<= 1) {
        for (int j = k >> 1; j > 0; j >>= 1) {
            if (t < NPIX) {
                int ixj = t ^ j;
                if (ixj > t) {
                    float a = sbuf[t], bb = sbuf[ixj];
                    bool up = ((t & k) == 0);
                    if (up ? (a > bb) : (a < bb)) { sbuf[t] = bb; sbuf[ixj] = a; }
                }
            }
            __syncthreads();
        }
    }

    if (t < NPIX) {
        int y = t / S, x = t - (t / S) * S;
        Ppad[(y + 1) * PS + (x + 1)] = sbuf[t];
    }
    __syncthreads();

    // conv1 (1->16) + folded BN + ReLU; two channels per iteration,
    // packed b32 store into channel-minor H1c (zero halo preserved).
    for (int idx = t; idx < 8 * NPIX; idx += NTH) {
        int p   = idx >> 3;
        int oc  = (idx & 7) * 2;
        int y   = p / S, x = p - (p / S) * S;
        float a0 = b1s[oc], a1 = b1s[oc + 1];
        #pragma unroll
        for (int ky = 0; ky < 3; ++ky)
            #pragma unroll
            for (int kx = 0; kx < 3; ++kx) {
                float pv = Ppad[(y + ky) * PS + (x + kx)];
                a0 += w1s[oc * 9 + ky * 3 + kx] * pv;
                a1 += w1s[(oc + 1) * 9 + ky * 3 + kx] * pv;
            }
        unsigned int packed = (unsigned int)f2bf(fmaxf(a0, 0.0f))
                            | ((unsigned int)f2bf(fmaxf(a1, 0.0f)) << 16);
        *(unsigned int*)&H1c[((y + 1) * PS + (x + 1)) * 16 + oc] = packed;
    }
    __syncthreads();

    // conv2 (16->32) as GEMM: M=32, N=NPIX, K'=144 (padded to 160 = 5 x 32)
    const int wv    = t >> 5;
    const int lane  = t & 31;
    const int col   = lane & 15;   // output pixel within N-tile
    const int khalf = lane >> 4;   // K' half supplied by this lane

    for (int task = wv; task < NTILE; task += NTH / 32) {
        int p = task * 16 + col;
        int y = p / S, x = p - (p / S) * S;
        const unsigned short* Hb = H1c + (y * PS + x) * 16; // halo in rowoff

        v8f acc0 = {};
        v8f acc1 = {};
        #pragma unroll
        for (int kt = 0; kt < 5; ++kt) {
            // this lane's tap q = kt*2 + khalf; q==9 lanes read a dummy
            // in-bounds row (their A entries are zero-padded).
            const int r0 = rowoff(kt * 2, PS) * 16;
            const int r1 = rowoff(kt * 2 + 1 > 8 ? 8 : kt * 2 + 1, PS) * 16;
            BU bu;
            bu.bv = *(const v16bf*)(Hb + (khalf ? r1 : r0)); // 32B LDS read
            v16bf a0 = *(const v16bf*)(afrag + ((kt)     * 32 + lane) * 16);
            v16bf a1 = *(const v16bf*)(afrag + ((5 + kt) * 32 + lane) * 16);
            acc0 = __builtin_amdgcn_wmma_f32_16x16x32_bf16(
                false, a0, false, bu.bv, (short)0, acc0, false, false);
            acc1 = __builtin_amdgcn_wmma_f32_16x16x32_bf16(
                false, a1, false, bu.bv, (short)0, acc1, false, false);
        }

        // bias + ReLU per element, sum over the 16 columns, LDS accumulate
        const int mb = khalf * 8;
        #pragma unroll
        for (int r = 0; r < 8; ++r) {
            float v0 = fmaxf(acc0[r] + b2s[mb + r], 0.0f);
            float v1 = fmaxf(acc1[r] + b2s[16 + mb + r], 0.0f);
            v0 += __shfl_xor(v0, 1, 32);
            v0 += __shfl_xor(v0, 2, 32);
            v0 += __shfl_xor(v0, 4, 32);
            v0 += __shfl_xor(v0, 8, 32);
            v1 += __shfl_xor(v1, 1, 32);
            v1 += __shfl_xor(v1, 2, 32);
            v1 += __shfl_xor(v1, 4, 32);
            v1 += __shfl_xor(v1, 8, 32);
            if (col == 0) {
                atomicAdd(&blockAcc[mb + r], v0);
                atomicAdd(&blockAcc[16 + mb + r], v1);
            }
        }
    }
    __syncthreads();

    // pixel-mean * block-mean == 1/768^2 for both block sizes
    if (t < 32)
        atomicAdd(&featAcc[sidx * 256 + b * 32 + t],
                  blockAcc[t] * (1.0f / 589824.0f));
}

// ---------------------------------------------------------------------------
// Kernel 3: [8,64] x [128,64]^T + bias
// ---------------------------------------------------------------------------
__global__ void guardian_proj(const float* __restrict__ featAcc,
                              const float* __restrict__ proj_w,
                              const float* __restrict__ proj_b,
                              float* __restrict__ out) {
    int idx = blockIdx.x * 256 + threadIdx.x;
    if (idx >= 8 * 128) return;
    int b = idx >> 7, o = idx & 127;
    float s = proj_b[o];
    #pragma unroll
    for (int j = 0; j < 64; ++j) {
        float f = featAcc[(j >> 5) * 256 + b * 32 + (j & 31)];
        s += f * proj_w[o * 64 + j];
    }
    out[idx] = s;
}

extern "C" void kernel_launch(void* const* d_in, const int* in_sizes, int n_in,
                              void* d_out, int out_size, void* d_ws, size_t ws_size,
                              hipStream_t stream) {
    (void)in_sizes; (void)n_in; (void)out_size; (void)ws_size;

    const float* c = (const float*)d_in[0];
    auto fill = [&](int base) {
        EncPtrs e;
        e.w1  = (const float*)d_in[base + 0];
        e.b1  = (const float*)d_in[base + 1];
        e.g1  = (const float*)d_in[base + 2];
        e.be1 = (const float*)d_in[base + 3];
        e.m1  = (const float*)d_in[base + 4];
        e.v1  = (const float*)d_in[base + 5];
        e.w2  = (const float*)d_in[base + 6];
        e.b2  = (const float*)d_in[base + 7];
        e.g2  = (const float*)d_in[base + 8];
        e.be2 = (const float*)d_in[base + 9];
        e.m2  = (const float*)d_in[base + 10];
        e.v2  = (const float*)d_in[base + 11];
        return e;
    };
    EncPtrs e0 = fill(1);
    EncPtrs e1 = fill(13);
    const float* proj_w = (const float*)d_in[25];
    const float* proj_b = (const float*)d_in[26];

    float*          fws     = (float*)d_ws;
    unsigned short* afrag   = (unsigned short*)((char*)d_ws + 8192);
    float*          featAcc = fws;

    guardian_prep<<<1, 256, 0, stream>>>(e0, e1, fws, afrag);

    guardian_block_enc<8><<<8 * 96 * 96, 128, 0, stream>>>(
        c, fws + 512, fws + 512 + 160, fws + 512 + 192, afrag, featAcc, 0);

    guardian_block_enc<16><<<8 * 48 * 48, 256, 0, stream>>>(
        c, fws + 1024, fws + 1024 + 160, fws + 1024 + 192, afrag + 5120,
        featAcc, 1);

    guardian_proj<<<4, 256, 0, stream>>>(featAcc, proj_w, proj_b, (float*)d_out);
}
